// EEGNetGNNTEECN_24266565223117
// MI455X (gfx1250) — compile-verified, matrix operand order checked by reading, not run on hardware
//
#include <hip/hip_runtime.h>
#include <hip/hip_bf16.h>

typedef _Float16 h16;
typedef __attribute__((ext_vector_type(16))) _Float16 v16h;
typedef __attribute__((ext_vector_type(8)))  _Float16 v8h;
typedef __attribute__((ext_vector_type(8)))  float    v8f;

#define BATCH 128
#define CH    64
#define TLEN  1024
#define EMB   32
#define KW    64
#define TOPK  8
#define NNODE (BATCH*CH)      // 8192
#define NEDGE (NNODE*TOPK)    // 65536
#define O1    256             // 4*64
#define O2    64
#define CCHUNK 64             // K elements staged per LDS chunk in k_corr
#define NCHUNK (TLEN / CCHUNK)

// ---------- WMMA fragment loaders (CDNA5 16x16x32 f16 layouts) ----------
// A fragment (16x32, row-major source, rows M, cols K):
//   lane<16:  M=lane,    K = k0+{0..7} and k0+{16..23}
//   lane>=16: M=lane-16, K = k0+{8..15} and k0+{24..31}
__device__ __forceinline__ v16h load_afrag(const h16* __restrict__ base, int m0, int ld, int k0) {
    int lane = threadIdx.x & 31;
    int r    = m0 + (lane & 15);
    int kk   = k0 + ((lane >> 4) & 1) * 8;
    const h16* p = base + (size_t)r * ld + kk;
    v8h lo = *(const v8h*)(p);        // K kk..kk+7
    v8h hi = *(const v8h*)(p + 16);   // K kk+16..kk+23
    return __builtin_shufflevector(lo, hi, 0,1,2,3,4,5,6,7,8,9,10,11,12,13,14,15);
}

// B fragment from a KxN row-major matrix (weights):
__device__ __forceinline__ v16h load_bfrag_kxn(const h16* __restrict__ W, int ld, int k0, int n0) {
    int lane = threadIdx.x & 31;
    int n    = n0 + (lane & 15);
    int kk   = k0 + ((lane >> 4) & 1) * 8;
    v16h f;
#pragma unroll
    for (int j = 0; j < 8; ++j) {
        f[j]     = W[(size_t)(kk + j) * ld + n];
        f[j + 8] = W[(size_t)(kk + 16 + j) * ld + n];
    }
    return f;
}

// C/D store: VGPR r -> M = m0+r (lane<16) or m0+8+r (lane>=16), N = n0+lane%16
__device__ __forceinline__ void store_cfrag(float* __restrict__ C, int ld, int m0, int n0, v8f acc) {
    int lane = threadIdx.x & 31;
    int n    = n0 + (lane & 15);
    int mb   = m0 + ((lane >> 4) & 1) * 8;
#pragma unroll
    for (int r = 0; r < 8; ++r) C[(size_t)(mb + r) * ld + n] = acc[r];
}

__device__ __forceinline__ float eluf(float a) { return a > 0.f ? a : (__expf(a) - 1.f); }

// ---------- CDNA5 async global->LDS copy (ASYNCcnt path) ----------
__device__ __forceinline__ void async_copy_b128(unsigned lds_off, const h16* g) {
    asm volatile("global_load_async_to_lds_b128 %0, %1, off"
                 :: "v"(lds_off), "v"(g) : "memory");
}
__device__ __forceinline__ void wait_async_le4() {
    asm volatile("s_wait_asynccnt 0x4" ::: "memory");
}
__device__ __forceinline__ void wait_async_0() {
    asm volatile("s_wait_asynccnt 0x0" ::: "memory");
}

// ---------- K1: per-row standardize -> f16 ----------
__global__ void k_norm(const float* __restrict__ x, h16* __restrict__ xs) {
    int row = blockIdx.x;
    const float* xr = x + (size_t)row * TLEN;
    __shared__ float r1[256], r2[256];
    float s = 0.f, sq = 0.f;
    for (int t = threadIdx.x; t < TLEN; t += 256) { float v = xr[t]; s += v; sq += v * v; }
    r1[threadIdx.x] = s; r2[threadIdx.x] = sq; __syncthreads();
    for (int st = 128; st > 0; st >>= 1) {
        if (threadIdx.x < st) { r1[threadIdx.x] += r1[threadIdx.x + st]; r2[threadIdx.x] += r2[threadIdx.x + st]; }
        __syncthreads();
    }
    float mean = r1[0] * (1.f / TLEN);
    float var  = (r2[0] - (float)TLEN * mean * mean) * (1.f / (TLEN - 1));
    float inv  = 1.f / (sqrtf(fmaxf(var, 0.f)) + 1e-8f);
    for (int t = threadIdx.x; t < TLEN; t += 256)
        xs[(size_t)row * TLEN + t] = (h16)((xr[t] - mean) * inv);
}

// ---------- K2: batched corr = xs_b * xs_b^T, async-LDS double buffered WMMA ----------
__global__ void k_corr(const h16* __restrict__ xs, float* __restrict__ corr) {
    int b    = blockIdx.x;              // 128 batches
    int tid  = threadIdx.x;             // 128 threads = 4 waves
    int wave = tid >> 5;                // each wave owns 16 rows x 64 cols
    __shared__ __align__(16) h16 sbuf[2][CH * CCHUNK];   // 2 x 8KB
    const h16* Xb = xs + (size_t)b * CH * TLEN;
    int m0 = wave * 16;

    // copy slice owned by this thread: row r, half p (32 halves = 4 x b128)
    int r = tid >> 1, p = tid & 1;
    const h16* gsrc0 = Xb + (size_t)r * TLEN + p * 32;
    unsigned lbase[2];
    lbase[0] = (unsigned)(size_t)(void*)&sbuf[0][0];
    lbase[1] = (unsigned)(size_t)(void*)&sbuf[1][0];
    unsigned lrow = (unsigned)((r * CCHUNK + p * 32) * 2);

    // prefetch chunk 0
#pragma unroll
    for (int q = 0; q < 4; ++q)
        async_copy_b128(lbase[0] + lrow + q * 16, gsrc0 + q * 8);

    v8f acc[4];
#pragma unroll
    for (int j = 0; j < 4; ++j)
#pragma unroll
        for (int t = 0; t < 8; ++t) acc[j][t] = 0.f;

    int buf = 0;
    for (int cc = 0; cc < NCHUNK; ++cc) {
        if (cc + 1 < NCHUNK) {
            const h16* gn = gsrc0 + (size_t)(cc + 1) * CCHUNK;
#pragma unroll
            for (int q = 0; q < 4; ++q)
                async_copy_b128(lbase[buf ^ 1] + lrow + q * 16, gn + q * 8);
            wait_async_le4();           // current chunk done (async ops in order)
        } else {
            wait_async_0();
        }
        __syncthreads();
        const h16* S = &sbuf[buf][0];
#pragma unroll
        for (int kk = 0; kk < CCHUNK; kk += 32) {
            v16h a = load_afrag(S, m0, CCHUNK, kk);
#pragma unroll
            for (int j = 0; j < 4; ++j) {
                v16h bf = load_afrag(S, j * 16, CCHUNK, kk);
                acc[j] = __builtin_amdgcn_wmma_f32_16x16x32_f16(false, a, false, bf, (short)0, acc[j], false, false);
            }
        }
        __syncthreads();
        buf ^= 1;
    }

    const float sc = 1.0f / (TLEN - 1);
#pragma unroll
    for (int j = 0; j < 4; ++j) {
        v8f rr = acc[j];
#pragma unroll
        for (int q = 0; q < 8; ++q) rr[q] = fminf(1.f, fmaxf(-1.f, rr[q] * sc));
        store_cfrag(corr + (size_t)b * CH * CH, CH, m0, j * 16, rr);
    }
}

// ---------- K3: top-8 |corr| per row (no self), build edge list + per-dst inbox ----------
__global__ void k_topk(const float* __restrict__ corr, float* __restrict__ ew,
                       int* __restrict__ indeg, int* __restrict__ inbox) {
    int row  = blockIdx.x * 8 + (threadIdx.x >> 5);   // node id
    if (row >= NNODE) return;
    int b = row >> 6, c = row & 63;
    int lane = threadIdx.x & 31;
    const float* cr = corr + (size_t)row * CH;
    float v0 = cr[lane], v1 = cr[lane + 32];
    unsigned k0 = __float_as_uint(fabsf(v0));   // abs() >= 0 -> bits monotonic
    unsigned k1 = __float_as_uint(fabsf(v1));
    if (lane == c)       k0 = 0u;               // mask diagonal
    if (lane + 32 == c)  k1 = 0u;
    for (int t = 0; t < TOPK; ++t) {
        unsigned bk; unsigned bi;
        if (k0 >= k1) { bk = k0; bi = (unsigned)lane; } else { bk = k1; bi = (unsigned)(lane + 32); }
        for (int off = 16; off; off >>= 1) {
            unsigned ok = __shfl_xor(bk, off, 32);
            unsigned oi = __shfl_xor(bi, off, 32);
            if (ok > bk || (ok == bk && oi < bi)) { bk = ok; bi = oi; }
        }
        int col = (int)bi;
        if (col < 32)  { if (lane == col)      k0 = 0u; }
        else           { if (lane == col - 32) k1 = 0u; }
        if (lane == 0) {
            int e = row * TOPK + t;
            int d = b * CH + col;
            ew[e] = cr[col];
            int pos = atomicAdd(&indeg[d], 1);
            inbox[d * 64 + pos] = e;            // src = e>>3
        }
    }
}

// ---------- K4: frontend -> h (f16, 8192x32) ----------
__global__ void k_frontend(const float* __restrict__ x, const float* __restrict__ dwk,
                           const float* __restrict__ g1, const float* __restrict__ b1,
                           const float* __restrict__ m1, const float* __restrict__ v1,
                           const float* __restrict__ wpw, const float* __restrict__ g2,
                           const float* __restrict__ b2, const float* __restrict__ m2,
                           const float* __restrict__ v2, h16* __restrict__ hh) {
    int row = blockIdx.x, c = row & 63;
    __shared__ float xb[TLEN + 64];
    __shared__ float wk[KW];
    __shared__ float yp[256];
    __shared__ float part[256];
    const float* xr = x + (size_t)row * TLEN;
    for (int i = threadIdx.x; i < TLEN + 64; i += 256) {
        int t = i - 32;
        xb[i] = (t >= 0 && t < TLEN) ? xr[t] : 0.f;
    }
    if (threadIdx.x < KW) wk[threadIdx.x] = dwk[c * KW + threadIdx.x];
    __syncthreads();
    float s1 = g1[c] * rsqrtf(v1[c] + 1e-5f);
    float o1 = b1[c] - m1[c] * s1;
    float pool = 0.f;
    int t0 = threadIdx.x * 4;
#pragma unroll
    for (int q = 0; q < 4; ++q) {
        float a = 0.f;
        int t = t0 + q;
#pragma unroll 8
        for (int k = 0; k < KW; ++k) a = fmaf(xb[t + k], wk[k], a);
        pool += eluf(a * s1 + o1);
    }
    yp[threadIdx.x] = pool * 0.25f;
    __syncthreads();
    // pointwise conv (outer product with w_pw) + BN2 + ELU + mean over L=256
    int e = threadIdx.x & 31, chunk = threadIdx.x >> 5;
    float s2 = g2[e] * rsqrtf(v2[e] + 1e-5f);
    float w  = wpw[e] * s2;
    float o2 = b2[e] - m2[e] * s2;
    float acc = 0.f;
    for (int l = chunk * 32; l < chunk * 32 + 32; ++l) acc += eluf(yp[l] * w + o2);
    part[threadIdx.x] = acc;
    __syncthreads();
    if (threadIdx.x < 32) {
        float s = 0.f;
        for (int q = 0; q < 8; ++q) s += part[q * 32 + threadIdx.x];
        hh[(size_t)row * EMB + threadIdx.x] = (h16)(s * (1.f / 256.f));
    }
}

// ---------- cast f32 weights -> f16 ----------
__global__ void k_cast(const float* __restrict__ s, h16* __restrict__ d, int n) {
    int i = blockIdx.x * 256 + threadIdx.x;
    if (i < n) d[i] = (h16)s[i];
}

// ---------- dual WMMA GEMM: XL = A*WL + bL, XR = A*WR + bR (A shared) ----------
__global__ void k_dualgemm(const h16* __restrict__ A, const h16* __restrict__ WL,
                           const h16* __restrict__ WR, const float* __restrict__ bL,
                           const float* __restrict__ bR, float* __restrict__ XL,
                           float* __restrict__ XR, int Kdim, int N, int ntn) {
    int wid = (blockIdx.x * 256 + threadIdx.x) >> 5;
    int mt = wid / ntn, nt = wid - mt * ntn;
    int m0 = mt * 16, n0 = nt * 16;
    v8f cl, cr;
#pragma unroll
    for (int r = 0; r < 8; ++r) { cl[r] = 0.f; cr[r] = 0.f; }
    for (int k0 = 0; k0 < Kdim; k0 += 32) {
        v16h a  = load_afrag(A, m0, Kdim, k0);
        v16h fl = load_bfrag_kxn(WL, N, k0, n0);
        v16h fr = load_bfrag_kxn(WR, N, k0, n0);
        cl = __builtin_amdgcn_wmma_f32_16x16x32_f16(false, a, false, fl, (short)0, cl, false, false);
        cr = __builtin_amdgcn_wmma_f32_16x16x32_f16(false, a, false, fr, (short)0, cr, false, false);
    }
    int lane = threadIdx.x & 31;
    int n  = n0 + (lane & 15);
    int mb = m0 + ((lane >> 4) & 1) * 8;
    float vl = bL[n], vr = bR[n];
#pragma unroll
    for (int r = 0; r < 8; ++r) {
        XL[(size_t)(mb + r) * N + n] = cl[r] + vl;
        XR[(size_t)(mb + r) * N + n] = cr[r] + vr;
    }
}

// ---------- K6: GATv2 layer-1 aggregation (H=4, Cg=64) + ELU -> f16 ----------
__global__ void k_gat1(const float* __restrict__ XL, const float* __restrict__ XR,
                       const float* __restrict__ ew, const float* __restrict__ We,
                       const float* __restrict__ att, const float* __restrict__ bias,
                       const int* __restrict__ indeg, const int* __restrict__ inbox,
                       h16* __restrict__ H1h) {
    int node = blockIdx.x, tid = threadIdx.x;            // 256 threads
    __shared__ float xr[O1], we[O1], at[O1], red[O1];
    __shared__ float logit[64 * 4];
    __shared__ float lmax[4], den[4];
    int deg = indeg[node];
    xr[tid] = XR[(size_t)node * O1 + tid];
    we[tid] = We[tid];
    at[tid] = att[tid];
    __syncthreads();
    int h = tid >> 6;
    for (int e = 0; e < deg; ++e) {
        int eid = inbox[node * 64 + e];
        int src = eid >> 3;
        float w = ew[eid];
        float m = XL[(size_t)src * O1 + tid] + xr[tid] + w * we[tid];
        m = m > 0.f ? m : 0.2f * m;                      // leaky_relu 0.2
        red[tid] = m * at[tid];
        __syncthreads();
        for (int st = 32; st; st >>= 1) {
            if ((tid & 63) < st) red[tid] += red[tid + st];
            __syncthreads();
        }
        if ((tid & 63) == 0) logit[e * 4 + h] = red[tid];
        __syncthreads();
    }
    if (tid < 4) {
        float mx = -1e30f;
        for (int e = 0; e < deg; ++e) mx = fmaxf(mx, logit[e * 4 + tid]);
        float s = 0.f;
        for (int e = 0; e < deg; ++e) s += __expf(logit[e * 4 + tid] - mx);
        lmax[tid] = mx; den[tid] = s > 0.f ? s : 1.f;
    }
    __syncthreads();
    float acc = 0.f;
    for (int e = 0; e < deg; ++e) {
        int eid = inbox[node * 64 + e];
        int src = eid >> 3;
        float a = __expf(logit[e * 4 + h] - lmax[h]) / den[h];
        acc += a * XL[(size_t)src * O1 + tid];
    }
    H1h[(size_t)node * O1 + tid] = (h16)eluf(acc + bias[tid]);
}

// ---------- K8: GATv2 layer-2 aggregation (H=1, Cg=64) + ELU ----------
__global__ void k_gat2(const float* __restrict__ XL, const float* __restrict__ XR,
                       const float* __restrict__ ew, const float* __restrict__ We,
                       const float* __restrict__ att, const float* __restrict__ bias,
                       const int* __restrict__ indeg, const int* __restrict__ inbox,
                       float* __restrict__ H2) {
    int node = blockIdx.x, tid = threadIdx.x;            // 64 threads
    __shared__ float xr[O2], we[O2], at[O2], red[O2], logit[64];
    __shared__ float lm, dn;
    int deg = indeg[node];
    xr[tid] = XR[(size_t)node * O2 + tid];
    we[tid] = We[tid];
    at[tid] = att[tid];
    __syncthreads();
    for (int e = 0; e < deg; ++e) {
        int eid = inbox[node * 64 + e];
        int src = eid >> 3;
        float w = ew[eid];
        float m = XL[(size_t)src * O2 + tid] + xr[tid] + w * we[tid];
        m = m > 0.f ? m : 0.2f * m;
        red[tid] = m * at[tid];
        __syncthreads();
        for (int st = 32; st; st >>= 1) {
            if (tid < st) red[tid] += red[tid + st];
            __syncthreads();
        }
        if (tid == 0) logit[e] = red[0];
        __syncthreads();
    }
    if (tid == 0) {
        float mx = -1e30f;
        for (int e = 0; e < deg; ++e) mx = fmaxf(mx, logit[e]);
        float s = 0.f;
        for (int e = 0; e < deg; ++e) s += __expf(logit[e] - mx);
        lm = mx; dn = s > 0.f ? s : 1.f;
    }
    __syncthreads();
    float acc = 0.f;
    for (int e = 0; e < deg; ++e) {
        int eid = inbox[node * 64 + e];
        int src = eid >> 3;
        acc += __expf(logit[e] - lm) / dn * XL[(size_t)src * O2 + tid];
    }
    H2[(size_t)node * O2 + tid] = eluf(acc + bias[tid]);
}

// ---------- K9: mean pool + TEECN residual + classifier ----------
__global__ void k_head(const float* __restrict__ H2,
                       const float* fc1w, const float* fc1b, const float* fc2w, const float* fc2b,
                       const float* p1w, const float* p1b, const float* p2w, const float* p2b,
                       const float* a1, const float* a2,
                       const float* c1w, const float* c1b, const float* c2w, const float* c2b,
                       float* __restrict__ out) {
    int b = blockIdx.x, j = threadIdx.x;                 // 64 threads
    __shared__ float g[O2], t1[O2], t2[O2], gg[O2], r1[O2];
    float s = 0.f;
    for (int c = 0; c < CH; ++c) s += H2[((size_t)b * CH + c) * O2 + j];
    g[j] = s * (1.f / CH);
    __syncthreads();
    float u1 = fc1b[j], u2 = fc2b[j];
    for (int i = 0; i < O2; ++i) { u1 = fmaf(g[i], fc1w[i * O2 + j], u1); u2 = fmaf(g[i], fc2w[i * O2 + j], u2); }
    t1[j] = tanhf(u1);
    float t = tanhf(u2); t2[j] = t * t;
    __syncthreads();
    float q1 = p1b[j], q2 = p2b[j];
    for (int i = 0; i < O2; ++i) { q1 = fmaf(t1[i], p1w[i * O2 + j], q1); q2 = fmaf(t2[i], p2w[i * O2 + j], q2); }
    gg[j] = g[j] + q1 * a1[j] + q2 * a2[j];
    __syncthreads();
    float r = c1b[j];
    for (int i = 0; i < O2; ++i) r = fmaf(gg[i], c1w[i * O2 + j], r);
    r1[j] = fmaxf(r, 0.f);
    __syncthreads();
    if (j < 2) {
        float o = c2b[j];
        for (int i = 0; i < O2; ++i) o = fmaf(r1[i], c2w[i * 2 + j], o);
        out[b * 2 + j] = o;
    }
}

extern "C" void kernel_launch(void* const* d_in, const int* in_sizes, int n_in,
                              void* d_out, int out_size, void* d_ws, size_t ws_size,
                              hipStream_t stream) {
    const float* x    = (const float*)d_in[0];
    const float* dwk  = (const float*)d_in[1];
    const float* g1   = (const float*)d_in[2];
    const float* b1   = (const float*)d_in[3];
    const float* m1   = (const float*)d_in[4];
    const float* v1   = (const float*)d_in[5];
    const float* wpw  = (const float*)d_in[6];
    const float* g2   = (const float*)d_in[7];
    const float* b2   = (const float*)d_in[8];
    const float* m2   = (const float*)d_in[9];
    const float* v2   = (const float*)d_in[10];
    const float* Wl1  = (const float*)d_in[11];
    const float* bl1  = (const float*)d_in[12];
    const float* Wr1  = (const float*)d_in[13];
    const float* br1  = (const float*)d_in[14];
    const float* We1  = (const float*)d_in[15];
    const float* att1 = (const float*)d_in[16];
    const float* bias1= (const float*)d_in[17];
    const float* Wl2  = (const float*)d_in[18];
    const float* bl2  = (const float*)d_in[19];
    const float* Wr2  = (const float*)d_in[20];
    const float* br2  = (const float*)d_in[21];
    const float* We2  = (const float*)d_in[22];
    const float* att2 = (const float*)d_in[23];
    const float* bias2= (const float*)d_in[24];
    const float* fc1w = (const float*)d_in[25];
    const float* fc1b = (const float*)d_in[26];
    const float* fc2w = (const float*)d_in[27];
    const float* fc2b = (const float*)d_in[28];
    const float* p1w  = (const float*)d_in[29];
    const float* p1b  = (const float*)d_in[30];
    const float* p2w  = (const float*)d_in[31];
    const float* p2b  = (const float*)d_in[32];
    const float* a1   = (const float*)d_in[33];
    const float* a2   = (const float*)d_in[34];
    const float* c1w  = (const float*)d_in[35];
    const float* c1b  = (const float*)d_in[36];
    const float* c2w  = (const float*)d_in[37];
    const float* c2b  = (const float*)d_in[38];
    float* out = (float*)d_out;

    char* base = (char*)d_ws;
    size_t off = 0;
    auto alloc = [&](size_t bytes) { size_t o = off; off = (off + bytes + 255) & ~(size_t)255; return o; };

    h16*   xs    = (h16*)  (base + alloc((size_t)NNODE * TLEN * 2));
    float* corr  = (float*)(base + alloc((size_t)BATCH * CH * CH * 4));
    float* ew    = (float*)(base + alloc((size_t)NEDGE * 4));
    int*   indeg = (int*)  (base + alloc((size_t)NNODE * 4));
    int*   inbox = (int*)  (base + alloc((size_t)NNODE * 64 * 4));
    h16*   hh    = (h16*)  (base + alloc((size_t)NNODE * EMB * 2));
    h16*   wl1h  = (h16*)  (base + alloc((size_t)EMB * O1 * 2));
    h16*   wr1h  = (h16*)  (base + alloc((size_t)EMB * O1 * 2));
    h16*   wl2h  = (h16*)  (base + alloc((size_t)O1 * O2 * 2));
    h16*   wr2h  = (h16*)  (base + alloc((size_t)O1 * O2 * 2));
    float* xl1   = (float*)(base + alloc((size_t)NNODE * O1 * 4));
    float* xr1   = (float*)(base + alloc((size_t)NNODE * O1 * 4));
    h16*   h1h   = (h16*)  (base + alloc((size_t)NNODE * O1 * 2));
    float* xl2   = (float*)(base + alloc((size_t)NNODE * O2 * 4));
    float* xr2   = (float*)(base + alloc((size_t)NNODE * O2 * 4));
    float* h2    = (float*)(base + alloc((size_t)NNODE * O2 * 4));
    (void)ws_size; (void)n_in; (void)in_sizes; (void)out_size;

    hipMemsetAsync(indeg, 0, (size_t)NNODE * 4, stream);

    // weight casts
    k_cast<<<(EMB * O1 + 255) / 256, 256, 0, stream>>>(Wl1, wl1h, EMB * O1);
    k_cast<<<(EMB * O1 + 255) / 256, 256, 0, stream>>>(Wr1, wr1h, EMB * O1);
    k_cast<<<(O1 * O2 + 255) / 256, 256, 0, stream>>>(Wl2, wl2h, O1 * O2);
    k_cast<<<(O1 * O2 + 255) / 256, 256, 0, stream>>>(Wr2, wr2h, O1 * O2);

    // graph build
    k_norm<<<NNODE, 256, 0, stream>>>(x, xs);
    k_corr<<<BATCH, 128, 0, stream>>>(xs, corr);
    k_topk<<<NNODE / 8, 256, 0, stream>>>(corr, ew, indeg, inbox);

    // frontend
    k_frontend<<<NNODE, 256, 0, stream>>>(x, dwk, g1, b1, m1, v1, wpw, g2, b2, m2, v2, hh);

    // GAT layer 1: linears (WMMA) + aggregation
    k_dualgemm<<<(NNODE / 16) * (O1 / 16) / 8, 256, 0, stream>>>(hh, wl1h, wr1h, bl1, br1, xl1, xr1, EMB, O1, O1 / 16);
    k_gat1<<<NNODE, 256, 0, stream>>>(xl1, xr1, ew, We1, att1, bias1, indeg, inbox, h1h);

    // GAT layer 2
    k_dualgemm<<<(NNODE / 16) * (O2 / 16) / 8, 256, 0, stream>>>(h1h, wl2h, wr2h, bl2, br2, xl2, xr2, O1, O2, O2 / 16);
    k_gat2<<<NNODE, 64, 0, stream>>>(xl2, xr2, ew, We2, att2, bias2, indeg, inbox, h2);

    // pool + head
    k_head<<<BATCH, 64, 0, stream>>>(h2, fc1w, fc1b, fc2w, fc2b, p1w, p1b, p2w, p2b,
                                     a1, a2, c1w, c1b, c2w, c2b, out);
}
